// LeWinTransformerBlock_39496519254585
// MI455X (gfx1250) — compile-verified, matrix operand order checked by reading, not run on hardware
//
#include <hip/hip_runtime.h>
#include <hip/hip_bf16.h>

typedef __bf16 bf16_t;
typedef __attribute__((ext_vector_type(16))) __bf16 v16bf;
typedef __attribute__((ext_vector_type(8)))  __bf16 v8bf;
typedef __attribute__((ext_vector_type(8)))  float  v8f;
typedef __attribute__((ext_vector_type(4)))  unsigned int u32x4;
typedef __attribute__((ext_vector_type(8)))  int i32x8;
typedef __attribute__((ext_vector_type(4)))  int i32x4;

#define TOK      131072                 // B*H*W tokens
#define NWIN     2048                   // total windows (B * 256)
#define SCALE_Q  0.17677669529663687f   // 32^-0.5

#if __has_builtin(__builtin_amdgcn_tensor_load_to_lds)
#define HAVE_TDM 1
#else
#define HAVE_TDM 0
#endif

union V16U { v16bf v; v8bf h[2]; };

// Load a 16x32 bf16 operand tile (A layout; B uses the same pattern on Bt rows).
// lane 0-15 : row = lane,    k-chunks {0..7, 16..23}
// lane 16-31: row = lane-16, k-chunks {8..15, 24..31}
__device__ __forceinline__ v16bf load_ab_tile(const bf16_t* __restrict__ base, int ld, int lane) {
  const int r  = lane & 15;
  const int ko = (lane >> 4) << 3;          // 0 or 8
  const bf16_t* p = base + r * ld + ko;
  V16U u;
  u.h[0] = *(const v8bf*)(p);               // 16B -> b128
  u.h[1] = *(const v8bf*)(p + 16);          // 16B -> b128
  return u.v;
}

__device__ __forceinline__ v8f zero8() {
  v8f z = {0.f,0.f,0.f,0.f,0.f,0.f,0.f,0.f};
  return z;
}

__device__ __forceinline__ float wave_sum32(float v) {
#pragma unroll
  for (int m = 1; m < 32; m <<= 1) v += __shfl_xor(v, m, 32);
  return v;
}

__device__ __forceinline__ int sregion(int h) {   // shift-mask region id
  return (h < 120) ? 0 : ((h < 124) ? 1 : 2);
}

#if HAVE_TDM
// TDM: DMA a [rows x kc] bf16 tile (global row stride ldk elems) into LDS
// (packed rows, contiguous). D# per CDNA5 ISA 08_async_tensor §8.3/8.4.
__device__ __forceinline__ void tdm_load_panel(const bf16_t* gsrc, unsigned lds_off,
                                               int rows, int kc, int ldk) {
  unsigned long long ga = (unsigned long long)(size_t)gsrc;
  u32x4 g0 = {
    1u,                                              // count=1 (valid user D#)
    lds_off,                                         // lds_addr (bytes)
    (unsigned)(ga & 0xFFFFFFFFu),                    // global_addr[31:0]
    (unsigned)((ga >> 32) & 0x01FFFFFFu) | (2u<<30)  // global_addr[56:32] | type=2
  };
  i32x8 g1 = {
    (int)(1u << 16),                                 // data_size=1 (2 bytes/elem)
    (int)((unsigned)kc << 16),                       // tensor_dim0[15:0] @bits63:48
    (int)((unsigned)rows << 16),                     // tensor_dim1[15:0] @bits95:80
    (int)((unsigned)kc << 16),                       // tile_dim0 @bits127:112
    rows,                                            // tile_dim1 @bits143:128
    ldk,                                             // tensor_dim0_stride[31:0]
    0, 0
  };
  i32x4 z4 = {0,0,0,0};
#if __clang_major__ >= 23
  i32x8 z8 = {0,0,0,0,0,0,0,0};
  __builtin_amdgcn_tensor_load_to_lds(g0, g1, z4, z4, z8, 0);
#else
  __builtin_amdgcn_tensor_load_to_lds(g0, g1, z4, z4, 0);
#endif
}
#endif

// Block-wide: stage Bt panel [64 x 256] (row stride ldk) into Bs, then each of
// the 8 waves accumulates its own 16x64 strip (rows from A, all cols of panel).
template <int K>
__device__ __forceinline__ void gemm_block(const bf16_t* __restrict__ Awave, // + m0*K
                                           const bf16_t* __restrict__ BtPanel,
                                           bf16_t* Bs, int wid, int lane, v8f acc[4]) {
#pragma unroll
  for (int kc0 = 0; kc0 < K; kc0 += 256) {
    __syncthreads();                      // previous panel fully consumed
#if HAVE_TDM
    if (wid == 0) {
      tdm_load_panel(BtPanel + kc0, (unsigned)(size_t)(void*)Bs, 64, 256, K);
      __builtin_amdgcn_s_wait_tensorcnt(0);
    }
#else
    // cooperative fallback: 256 threads * 16B chunks
    for (int idx = threadIdx.x; idx < (64 * 256) / 8; idx += 256) {
      int rr = idx >> 5, cc = (idx & 31) << 3;
      *(v8bf*)(Bs + rr * 256 + cc) = *(const v8bf*)(BtPanel + kc0 + (size_t)rr * K + cc);
    }
#endif
    __syncthreads();

    v16bf a[8];
#pragma unroll
    for (int s = 0; s < 8; ++s) {
      a[s] = load_ab_tile(Awave + kc0 + s * 32, K, lane);
      __builtin_prefetch(Awave + kc0 + s * 32 + 256, 0, 0);
    }
#pragma unroll
    for (int s = 0; s < 8; ++s)
#pragma unroll
      for (int j = 0; j < 4; ++j) {
        v16bf b = load_ab_tile(Bs + j * 16 * 256 + s * 32, 256, lane);
        acc[j] = __builtin_amdgcn_wmma_f32_16x16x32_bf16(false, a[s], false, b,
                                                         (short)0, acc[j], false, false);
      }
  }
}

// ---------------------------------------------------------------- weights pack
__global__ void k_pack_weights(const float* __restrict__ wq,  const float* __restrict__ wkv,
                               const float* __restrict__ pw,  const float* __restrict__ f1,
                               const float* __restrict__ f2,
                               bf16_t* wqT, bf16_t* wkvT, bf16_t* projT,
                               bf16_t* fc1T, bf16_t* fc2T) {
  int i = blockIdx.x * blockDim.x + threadIdx.x;     // 262144 threads
  if (i < 65536) {
    int o = i >> 8, in = i & 255;
    wqT  [o * 256 + in] = (bf16_t)wq[in * 256 + o];
    projT[o * 256 + in] = (bf16_t)pw[in * 256 + o];
  }
  if (i < 131072) {
    int o = i >> 8, in = i & 255;
    wkvT[o * 256 + in] = (bf16_t)wkv[in * 512 + o];
  }
  if (i < 262144) {
    int o = i >> 8, in = i & 255;
    fc1T[o * 256 + in] = (bf16_t)f1[in * 1024 + o];
    int o2 = i >> 10, in2 = i & 1023;
    fc2T[o2 * 1024 + in2] = (bf16_t)f2[in2 * 256 + o2];
  }
}

// ------------------------------------------ LN1 + cyclic shift + window gather
__global__ __launch_bounds__(256) void k_ln1_window(const float* __restrict__ x,
                                                    const float* __restrict__ g,
                                                    const float* __restrict__ bta,
                                                    bf16_t* __restrict__ xw) {
  int token = blockIdx.x * 8 + (threadIdx.x >> 5);
  int lane  = threadIdx.x & 31;
  int n  = token & 63, win = token >> 6;
  int ww = win & 15, wh = (win >> 4) & 15, b = win >> 8;
  int yi = n >> 3, xj = n & 7;
  int sh = (wh * 8 + yi + 4) & 127;                  // roll(-SHIFT) gather
  int sw = (ww * 8 + xj + 4) & 127;
  const float* row = x + ((size_t)b * 16384 + sh * 128 + sw) * 256;
  float v[8]; float s = 0.f;
#pragma unroll
  for (int u = 0; u < 8; ++u) { v[u] = row[lane * 8 + u]; s += v[u]; }
  float mean = wave_sum32(s) * (1.f / 256.f);
  float q = 0.f;
#pragma unroll
  for (int u = 0; u < 8; ++u) { float d = v[u] - mean; q += d * d; }
  float rstd = rsqrtf(wave_sum32(q) * (1.f / 256.f) + 1e-5f);
  bf16_t* dst = xw + (size_t)token * 256;
#pragma unroll
  for (int u = 0; u < 8; ++u) {
    int c = lane * 8 + u;
    dst[c] = (bf16_t)((v[u] - mean) * rstd * g[c] + bta[c]);
  }
}

// ------------------------------------------------------------------- QKV GEMM
__global__ __launch_bounds__(256) void k_gemm_qkv(const bf16_t* __restrict__ A,
                                                  const bf16_t* __restrict__ wqT,
                                                  const bf16_t* __restrict__ wkvT,
                                                  const float* __restrict__ bq,
                                                  const float* __restrict__ bkv,
                                                  bf16_t* __restrict__ Q,
                                                  bf16_t* __restrict__ Kk,
                                                  bf16_t* __restrict__ Vt) {
  __shared__ bf16_t Bs[64 * 256];
  int wid = threadIdx.x >> 5, lane = threadIdx.x & 31;
  int mb = blockIdx.x / 12, ns = blockIdx.x % 12;
  int m0 = mb * 128 + wid * 16, c0 = ns * 64;
  const bf16_t* Bt = (c0 < 256) ? (wqT + (size_t)c0 * 256)
                                : (wkvT + (size_t)(c0 - 256) * 256);
  v8f acc[4];
#pragma unroll
  for (int j = 0; j < 4; ++j) acc[j] = zero8();
  gemm_block<256>(A + (size_t)m0 * 256, Bt, Bs, wid, lane, acc);

  int col = lane & 15, half = lane >> 4;
#pragma unroll
  for (int j = 0; j < 4; ++j)
#pragma unroll
    for (int r = 0; r < 8; ++r) {
      int m = m0 + r + 8 * half;
      int c = c0 + j * 16 + col;
      float val = acc[j][r];
      int wi = m >> 6, nn = m & 63;
      if (c < 256) {                                            // q (pre-scaled)
        val = (val + bq[c]) * SCALE_Q;
        int hh = c >> 5, hd = c & 31;
        Q[(((size_t)wi * 8 + hh) * 64 + nn) * 32 + hd] = (bf16_t)val;
      } else {
        int kc = c - 256;
        val += bkv[kc];
        if (kc < 256) {                                         // k row-major
          int hh = kc >> 5, hd = kc & 31;
          Kk[(((size_t)wi * 8 + hh) * 64 + nn) * 32 + hd] = (bf16_t)val;
        } else {                                                // v transposed
          int vc = kc - 256, hh = vc >> 5, hd = vc & 31;
          Vt[(((size_t)wi * 8 + hh) * 32 + hd) * 64 + nn] = (bf16_t)val;
        }
      }
    }
}

// ---------------------------------------------------- attention (1 wave/head)
__global__ __launch_bounds__(256) void k_attn(const bf16_t* __restrict__ Q,
                                              const bf16_t* __restrict__ Kk,
                                              const bf16_t* __restrict__ Vt,
                                              const float* __restrict__ rpb,
                                              bf16_t* __restrict__ AO) {
  __shared__ bf16_t Pl[8][64 * 64];                  // 64 KB
  int wid = threadIdx.x >> 5, lane = threadIdx.x & 31;
  int win = blockIdx.x, h = wid;
  const bf16_t* qp = Q  + ((size_t)win * 8 + h) * 2048;
  const bf16_t* kp = Kk + ((size_t)win * 8 + h) * 2048;
  const bf16_t* vp = Vt + ((size_t)win * 8 + h) * 2048;

  v16bf a[4];
#pragma unroll
  for (int i = 0; i < 4; ++i) a[i] = load_ab_tile(qp + i * 512, 32, lane);

  v8f S[4][4];
#pragma unroll
  for (int i = 0; i < 4; ++i)
#pragma unroll
    for (int j = 0; j < 4; ++j) S[i][j] = zero8();

#pragma unroll
  for (int j = 0; j < 4; ++j) {
    v16bf bt = load_ab_tile(kp + j * 512, 32, lane);
#pragma unroll
    for (int i = 0; i < 4; ++i)
      S[i][j] = __builtin_amdgcn_wmma_f32_16x16x32_bf16(false, a[i], false, bt,
                                                        (short)0, S[i][j], false, false);
  }

  // relative-position bias + shift mask
  int col = lane & 15, half = lane >> 4;
  int wIn = win & 255, wh = wIn >> 4, ww = wIn & 15;
#pragma unroll
  for (int i = 0; i < 4; ++i)
#pragma unroll
    for (int j = 0; j < 4; ++j)
#pragma unroll
      for (int r = 0; r < 8; ++r) {
        int qn = i * 16 + r + 8 * half, km = j * 16 + col;
        int qy = qn >> 3, qx = qn & 7, ky = km >> 3, kx = km & 7;
        float bias = rpb[((qy - ky + 7) * 15 + (qx - kx + 7)) * 8 + h];
        int rq = sregion(wh * 8 + qy) * 3 + sregion(ww * 8 + qx);
        int rk = sregion(wh * 8 + ky) * 3 + sregion(ww * 8 + kx);
        S[i][j][r] += bias + ((rq != rk) ? -100.f : 0.f);
      }

  // row softmax (rows live across j-tiles + 16-lane groups)
#pragma unroll
  for (int i = 0; i < 4; ++i)
#pragma unroll
    for (int r = 0; r < 8; ++r) {
      float mx = fmaxf(fmaxf(S[i][0][r], S[i][1][r]), fmaxf(S[i][2][r], S[i][3][r]));
      mx = fmaxf(mx, __shfl_xor(mx, 1, 32));
      mx = fmaxf(mx, __shfl_xor(mx, 2, 32));
      mx = fmaxf(mx, __shfl_xor(mx, 4, 32));
      mx = fmaxf(mx, __shfl_xor(mx, 8, 32));
      float sum = 0.f;
#pragma unroll
      for (int j = 0; j < 4; ++j) { float e = __expf(S[i][j][r] - mx); S[i][j][r] = e; sum += e; }
      sum += __shfl_xor(sum, 1, 32);
      sum += __shfl_xor(sum, 2, 32);
      sum += __shfl_xor(sum, 4, 32);
      sum += __shfl_xor(sum, 8, 32);
      float inv = 1.f / sum;
#pragma unroll
      for (int j = 0; j < 4; ++j) S[i][j][r] *= inv;
    }

  // stage P (bf16, row-major) into this wave's LDS slice
  bf16_t* pl = &Pl[wid][0];
#pragma unroll
  for (int i = 0; i < 4; ++i)
#pragma unroll
    for (int j = 0; j < 4; ++j)
#pragma unroll
      for (int r = 0; r < 8; ++r)
        pl[(i * 16 + r + 8 * half) * 64 + j * 16 + col] = (bf16_t)S[i][j][r];

  // O = P * V   (A from LDS, B = V^T rows = head-dim channels)
  v8f O[4][2];
#pragma unroll
  for (int i = 0; i < 4; ++i) { O[i][0] = zero8(); O[i][1] = zero8(); }
#pragma unroll
  for (int ks = 0; ks < 64; ks += 32) {
    v16bf pa[4];
#pragma unroll
    for (int i = 0; i < 4; ++i) pa[i] = load_ab_tile(pl + i * 1024 + ks, 64, lane);
#pragma unroll
    for (int jh = 0; jh < 2; ++jh) {
      v16bf bt = load_ab_tile(vp + jh * 1024 + ks, 64, lane);
#pragma unroll
      for (int i = 0; i < 4; ++i)
        O[i][jh] = __builtin_amdgcn_wmma_f32_16x16x32_bf16(false, pa[i], false, bt,
                                                           (short)0, O[i][jh], false, false);
    }
  }

#pragma unroll
  for (int i = 0; i < 4; ++i)
#pragma unroll
    for (int jh = 0; jh < 2; ++jh)
#pragma unroll
      for (int r = 0; r < 8; ++r) {
        int tok = win * 64 + i * 16 + r + 8 * half;
        int cc  = h * 32 + jh * 16 + col;
        AO[(size_t)tok * 256 + cc] = (bf16_t)O[i][jh][r];
      }
}

// --------------------------- proj GEMM + window reverse + un-shift + residual
__global__ __launch_bounds__(256) void k_proj(const bf16_t* __restrict__ AO,
                                              const bf16_t* __restrict__ projT,
                                              const float* __restrict__ pb,
                                              const float* __restrict__ x,
                                              float* __restrict__ out) {
  __shared__ bf16_t Bs[64 * 256];
  int wid = threadIdx.x >> 5, lane = threadIdx.x & 31;
  int mb = blockIdx.x >> 2, ns = blockIdx.x & 3;
  int m0 = mb * 128 + wid * 16, c0 = ns * 64;
  v8f acc[4];
#pragma unroll
  for (int j = 0; j < 4; ++j) acc[j] = zero8();
  gemm_block<256>(AO + (size_t)m0 * 256, projT + (size_t)c0 * 256, Bs, wid, lane, acc);

  int col = lane & 15, half = lane >> 4;
#pragma unroll
  for (int j = 0; j < 4; ++j)
#pragma unroll
    for (int r = 0; r < 8; ++r) {
      int m = m0 + r + 8 * half;
      int c = c0 + j * 16 + col;
      int wi = m >> 6, n = m & 63;
      int b = wi >> 8, wh = (wi >> 4) & 15, ww = wi & 15;
      int yi = n >> 3, xj = n & 7;
      int dh = (wh * 8 + yi + 4) & 127;               // roll(+SHIFT) scatter
      int dw = (ww * 8 + xj + 4) & 127;
      size_t t = (size_t)b * 16384 + dh * 128 + dw;
      out[t * 256 + c] = x[t * 256 + c] + acc[j][r] + pb[c];
    }
}

// ------------------------------------------------------------------------ LN2
__global__ __launch_bounds__(256) void k_ln2(const float* __restrict__ x2,
                                             const float* __restrict__ g,
                                             const float* __restrict__ bta,
                                             bf16_t* __restrict__ xm) {
  int token = blockIdx.x * 8 + (threadIdx.x >> 5);
  int lane  = threadIdx.x & 31;
  const float* row = x2 + (size_t)token * 256;
  float v[8]; float s = 0.f;
#pragma unroll
  for (int u = 0; u < 8; ++u) { v[u] = row[lane * 8 + u]; s += v[u]; }
  float mean = wave_sum32(s) * (1.f / 256.f);
  float q = 0.f;
#pragma unroll
  for (int u = 0; u < 8; ++u) { float d = v[u] - mean; q += d * d; }
  float rstd = rsqrtf(wave_sum32(q) * (1.f / 256.f) + 1e-5f);
  bf16_t* dst = xm + (size_t)token * 256;
#pragma unroll
  for (int u = 0; u < 8; ++u) {
    int c = lane * 8 + u;
    dst[c] = (bf16_t)((v[u] - mean) * rstd * g[c] + bta[c]);
  }
}

// ------------------------------------------------------------- FC1 + GELU
__global__ __launch_bounds__(256) void k_fc1(const bf16_t* __restrict__ Xm,
                                             const bf16_t* __restrict__ fc1T,
                                             const float* __restrict__ b1,
                                             bf16_t* __restrict__ Hd) {
  __shared__ bf16_t Bs[64 * 256];
  int wid = threadIdx.x >> 5, lane = threadIdx.x & 31;
  int mb = blockIdx.x >> 4, ns = blockIdx.x & 15;
  int m0 = mb * 128 + wid * 16, c0 = ns * 64;
  v8f acc[4];
#pragma unroll
  for (int j = 0; j < 4; ++j) acc[j] = zero8();
  gemm_block<256>(Xm + (size_t)m0 * 256, fc1T + (size_t)c0 * 256, Bs, wid, lane, acc);

  int col = lane & 15, half = lane >> 4;
#pragma unroll
  for (int j = 0; j < 4; ++j)
#pragma unroll
    for (int r = 0; r < 8; ++r) {
      int m = m0 + r + 8 * half;
      int c = c0 + j * 16 + col;
      float v = acc[j][r] + b1[c];
      float gl = 0.5f * v * (1.f + erff(v * 0.70710678118654752f));  // exact GELU
      Hd[(size_t)m * 1024 + c] = (bf16_t)gl;
    }
}

// ------------------------------------------------------ FC2 + final residual
__global__ __launch_bounds__(256) void k_fc2(const bf16_t* __restrict__ Hd,
                                             const bf16_t* __restrict__ fc2T,
                                             const float* __restrict__ b2,
                                             float* __restrict__ out) {
  __shared__ bf16_t Bs[64 * 256];
  int wid = threadIdx.x >> 5, lane = threadIdx.x & 31;
  int mb = blockIdx.x >> 2, ns = blockIdx.x & 3;
  int m0 = mb * 128 + wid * 16, c0 = ns * 64;
  v8f acc[4];
#pragma unroll
  for (int j = 0; j < 4; ++j) acc[j] = zero8();
  gemm_block<1024>(Hd + (size_t)m0 * 1024, fc2T + (size_t)c0 * 1024, Bs, wid, lane, acc);

  int col = lane & 15, half = lane >> 4;
#pragma unroll
  for (int j = 0; j < 4; ++j)
#pragma unroll
    for (int r = 0; r < 8; ++r) {
      int m = m0 + r + 8 * half;
      int c = c0 + j * 16 + col;
      out[(size_t)m * 256 + c] += acc[j][r] + b2[c];
    }
}

// ------------------------------------------------------------------ launcher
extern "C" void kernel_launch(void* const* d_in, const int* in_sizes, int n_in,
                              void* d_out, int out_size, void* d_ws, size_t ws_size,
                              hipStream_t stream) {
  (void)in_sizes; (void)n_in; (void)out_size; (void)ws_size;

  const float* x       = (const float*)d_in[0];
  const float* n1g     = (const float*)d_in[1];
  const float* n1b     = (const float*)d_in[2];
  const float* wq      = (const float*)d_in[3];
  const float* bq      = (const float*)d_in[4];
  const float* wkv     = (const float*)d_in[5];
  const float* bkv     = (const float*)d_in[6];
  const float* rpb     = (const float*)d_in[7];
  const float* proj_w  = (const float*)d_in[8];
  const float* proj_b  = (const float*)d_in[9];
  const float* n2g     = (const float*)d_in[10];
  const float* n2b     = (const float*)d_in[11];
  const float* fc1_w   = (const float*)d_in[12];
  const float* fc1_b   = (const float*)d_in[13];
  const float* fc2_w   = (const float*)d_in[14];
  const float* fc2_b   = (const float*)d_in[15];
  float* out = (float*)d_out;

  char* ws = (char*)d_ws;
  bf16_t* wqT   = (bf16_t*)(ws);                          // 128 KB
  bf16_t* wkvT  = (bf16_t*)(ws + 131072);                 // 256 KB
  bf16_t* projT = (bf16_t*)(ws + 393216);                 // 128 KB
  bf16_t* fc1T  = (bf16_t*)(ws + 524288);                 // 512 KB
  bf16_t* fc2T  = (bf16_t*)(ws + 1048576);                // 512 KB
  bf16_t* Abuf  = (bf16_t*)(ws + (2ull  << 20));          // 64 MB (xw, later xm)
  bf16_t* Qb    = (bf16_t*)(ws + (70ull << 20));          // 64 MB
  bf16_t* Kb    = Qb  + 33554432;                         // 64 MB
  bf16_t* Vb    = Kb  + 33554432;                         // 64 MB
  bf16_t* AOb   = Vb  + 33554432;                         // 64 MB
  bf16_t* Hb    = (bf16_t*)(ws + (70ull << 20));          // 256 MB (reuses Q..AO)

  k_pack_weights<<<1024, 256, 0, stream>>>(wq, wkv, proj_w, fc1_w, fc2_w,
                                           wqT, wkvT, projT, fc1T, fc2T);
  k_ln1_window<<<TOK / 8, 256, 0, stream>>>(x, n1g, n1b, Abuf);
  k_gemm_qkv<<<1024 * 12, 256, 0, stream>>>(Abuf, wqT, wkvT, bq, bkv, Qb, Kb, Vb);
  k_attn<<<NWIN, 256, 0, stream>>>(Qb, Kb, Vb, rpb, AOb);
  k_proj<<<1024 * 4, 256, 0, stream>>>(AOb, projT, proj_b, x, out);
  k_ln2<<<TOK / 8, 256, 0, stream>>>(out, n2g, n2b, Abuf);
  k_fc1<<<1024 * 16, 256, 0, stream>>>(Abuf, fc1T, fc1_b, Hb);
  k_fc2<<<1024 * 4, 256, 0, stream>>>(Hb, fc2T, fc2_b, out);
}
